// GraphConvBlock_30099130811042
// MI455X (gfx1250) — matrix operand order repaired and checked by reading, not compiled, not run on hardware
//
#include <hip/hip_runtime.h>
#include <hip/hip_bf16.h>

// GraphConvBlock for MI455X (gfx1250, wave32, WMMA).
// msg = P @ W2r with P[e, j*64+h] = hr[e,j]*x_src[e,h]:  M=65536 K=4096 N=64 (~34.4 GFLOP)
// -> bf16 WMMA / f32 accum. A fragments generated in-register with V_PK_MUL_BF16;
// x fragments hoisted (only 2 distinct K-slices per row); hr scalars loaded 8-at-a-time.

namespace {

constexpr int NN = 32768;   // nodes
constexpr int NE = 65536;   // edges
constexpr int HD = 64;      // hidden
constexpr float LNEPS = 1e-5f;

typedef __attribute__((ext_vector_type(16))) __bf16 v16bf;
typedef __attribute__((ext_vector_type(8)))  float  v8f;

union BV {
  v16bf v;
  unsigned short u[16];
  uint4 q[2];
};

union HB {            // 8 packed bf16 scalars
  uint4 q;
  __bf16 b[8];
};

__device__ __forceinline__ unsigned short f2bf(float f) {
  unsigned u = __float_as_uint(f);
  u += 0x7FFFu + ((u >> 16) & 1u);   // round-to-nearest-even
  return (unsigned short)(u >> 16);
}
__device__ __forceinline__ float bf2f(unsigned short h) {
  return __uint_as_float(((unsigned)h) << 16);
}

__device__ __forceinline__ v8f wmma_bf16(const BV& a, const BV& b, v8f c) {
  return __builtin_amdgcn_wmma_f32_16x16x32_bf16(false, a.v, false, b.v,
                                                 (short)0, c, false, false);
}

__device__ __forceinline__ v16bf splat16(__bf16 s) {
  v16bf r;
#pragma unroll
  for (int i = 0; i < 16; ++i) r[i] = s;
  return r;
}

// ---------------- prep kernels ----------------

__global__ void k_zero(float* agg, float* deg) {
  int t = blockIdx.x * blockDim.x + threadIdx.x;
  if (t < NN * HD) agg[t] = 0.0f;
  if (t < NN) deg[t] = 0.0f;
}

// Bbuf[n][j*64+h] = bf16( W2[j][h*64+n] )  -> B matrix of the big GEMM, n-major
__global__ void k_prep_w2(const float* __restrict__ W2, unsigned short* __restrict__ Bbuf) {
  int t = blockIdx.x * blockDim.x + threadIdx.x;   // 64*64*64
  int n = t >> 12, j = (t >> 6) & 63, h = t & 63;
  Bbuf[n * 4096 + j * 64 + h] = f2bf(W2[j * 4096 + h * 64 + n]);
}

// small 64x64 transposes into n-major bf16
__global__ void k_prep_small(const float* __restrict__ W1, const float* __restrict__ b2,
                             const float* __restrict__ root,
                             unsigned short* __restrict__ W1c,
                             unsigned short* __restrict__ Bbias,
                             unsigned short* __restrict__ rootc) {
  int t = blockIdx.x * blockDim.x + threadIdx.x;   // 4096
  int n = t >> 6, k = t & 63;
  W1c[n * 64 + k]   = f2bf(W1[k * 64 + n]);
  Bbias[n * 64 + k] = f2bf(b2[k * 64 + n]);        // Bbias[n][h] = b2[h*64+n]
  rootc[n * 64 + k] = f2bf(root[k * 64 + n]);
}

__global__ void k_gather(const float* __restrict__ node, const long long* __restrict__ src,
                         unsigned short* __restrict__ Xsrc) {
  int t = blockIdx.x * blockDim.x + threadIdx.x;   // NE*64
  int e = t >> 6, h = t & 63;
  long long s = src[e];
  Xsrc[t] = f2bf(node[(int)s * HD + h]);
}

__global__ void k_deg(const long long* __restrict__ dst, float* deg) {
  int t = blockIdx.x * blockDim.x + threadIdx.x;
  if (t < NE) atomicAdd(&deg[(int)dst[t]], 1.0f);
}

// ---------------- hr = relu(edge_emb @ W1 + b1), bf16 out ----------------
// 128 threads = 4 waves; 16 edges / block; wave w -> N-tile w*16.
__global__ void k_hr(const float* __restrict__ edge_emb, const unsigned short* __restrict__ W1c,
                     const float* __restrict__ b1, unsigned short* __restrict__ hr) {
  __shared__ unsigned short ae[16 * 64];
  const int e0 = blockIdx.x * 16;
  const int tid = threadIdx.x;
  for (int i = tid; i < 16 * 64; i += 128) ae[i] = f2bf(edge_emb[e0 * 64 + i]);
  __syncthreads();

  const int w = tid >> 5, l = tid & 31;
  const int m = l & 15, g = l >> 4;
  const int n = w * 16 + m;         // output column of this lane (B/C/D layout)
  v8f acc = {};
#pragma unroll
  for (int kc = 0; kc < 2; ++kc) {
    const int xoff = kc * 32;
    BV A, B;
    const unsigned short* ar = &ae[m * 64 + xoff];
#pragma unroll
    for (int i = 0; i < 8; ++i) {       // A 16-bit 16x32 layout: K = 8g+i and 16+8g+i
      A.u[i]     = ar[g * 8 + i];
      A.u[8 + i] = ar[16 + g * 8 + i];
    }
    const uint4* bp = (const uint4*)&W1c[n * 64 + xoff + g * 8];  // B: K = 8g..8g+7, 16+8g..16+8g+7
    B.q[0] = bp[0]; B.q[1] = bp[2];
    acc = wmma_bf16(A, B, acc);
  }
#pragma unroll
  for (int r = 0; r < 8; ++r) {
    const int M = r + 8 * g;            // C/D layout
    float v = acc[r] + b1[n];
    hr[(e0 + M) * 64 + n] = f2bf(v > 0.0f ? v : 0.0f);
  }
}

// ---------------- main: msg = (hr (x) x_src) @ W2r + x_src @ Bbias; scatter-add ----------------
// 128 threads = 4 waves; 64 edges / block; each wave: four 16x16 C tiles (rows 0-15..48-63),
// columns w*16..w*16+15.  B fragment loaded once per K-chunk, reused by all 4 tiles.
__global__ void k_msg(const unsigned short* __restrict__ hr, const unsigned short* __restrict__ Xsrc,
                      const unsigned short* __restrict__ Bbuf, const unsigned short* __restrict__ Bbias,
                      const long long* __restrict__ dst, float* __restrict__ agg) {
  __shared__ int ds_s[64];
  const int e0 = blockIdx.x * 64;
  const int tid = threadIdx.x;
  if (tid < 64) ds_s[tid] = (int)dst[e0 + tid];

  const int w = tid >> 5, l = tid & 31;
  const int m = l & 15, g = l >> 4;
  const int n = w * 16 + m;

  // Hoist x fragments: tile t (edges e0+16t+m), K-slice p (0..31 / 32..63).
  // A-fragment layout: elems 0-7 -> K = 32p + 8g + i, elems 8-15 -> K = 32p + 16 + 8g + i.
  BV xf[4][2];
#pragma unroll
  for (int t = 0; t < 4; ++t) {
#pragma unroll
    for (int p = 0; p < 2; ++p) {
      const unsigned short* xr = Xsrc + (e0 + 16 * t + m) * 64 + 32 * p + 8 * g;
      xf[t][p].q[0] = *(const uint4*)xr;
      xf[t][p].q[1] = *(const uint4*)(xr + 16);
    }
  }

  v8f acc[4];
#pragma unroll
  for (int t = 0; t < 4; ++t) acc[t] = (v8f){};

  // bias term: x_src @ Bbias  (K = 64 -> 2 chunks), accumulator init
#pragma unroll
  for (int p = 0; p < 2; ++p) {
    BV B;
    const uint4* bp = (const uint4*)&Bbias[n * 64 + 32 * p + g * 8];
    B.q[0] = bp[0]; B.q[1] = bp[2];
#pragma unroll
    for (int t = 0; t < 4; ++t) acc[t] = wmma_bf16(xf[t][p], B, acc[t]);
  }

  // main K loop: 4096 = 64 j-planes x 2 slices.  j-scalars fetched 8 at a time (b128).
  for (int jo = 0; jo < 8; ++jo) {
    HB hq[4];
#pragma unroll
    for (int t = 0; t < 4; ++t)
      hq[t].q = *(const uint4*)&hr[(e0 + 16 * t + m) * 64 + jo * 8];
    if (jo < 7)
      __builtin_prefetch((const void*)&Bbuf[n * 4096 + (jo + 1) * 8 * 64], 0, 1);
#pragma unroll
    for (int ji = 0; ji < 8; ++ji) {
      const int j = jo * 8 + ji;
      v16bf sv[4];
#pragma unroll
      for (int t = 0; t < 4; ++t) sv[t] = splat16(hq[t].b[ji]);
#pragma unroll
      for (int p = 0; p < 2; ++p) {
        BV B;
        const uint4* bp = (const uint4*)&Bbuf[n * 4096 + j * 64 + 32 * p + g * 8];
        B.q[0] = bp[0]; B.q[1] = bp[2];
#pragma unroll
        for (int t = 0; t < 4; ++t) {
          BV A;
          A.v = xf[t][p].v * sv[t];          // V_PK_MUL_BF16 x8
          acc[t] = wmma_bf16(A, B, acc[t]);
        }
      }
    }
  }

  __syncthreads();
#pragma unroll
  for (int t = 0; t < 4; ++t) {
#pragma unroll
    for (int r = 0; r < 8; ++r) {
      const int M = r + 8 * g;
      atomicAdd(&agg[ds_s[16 * t + M] * HD + n], acc[t][r]);
    }
  }
}

// ---------------- out = LN( agg/deg + node@root + bias + node ) ----------------
__global__ void k_out(const float* __restrict__ node, const unsigned short* __restrict__ rootc,
                      const float* __restrict__ bias, const float* __restrict__ gamma,
                      const float* __restrict__ beta, const float* __restrict__ agg,
                      const float* __restrict__ deg, float* __restrict__ out) {
  __shared__ unsigned short xn[16 * 64];
  __shared__ float ob[16 * 64];
  __shared__ float mu_s[16], rs_s[16];
  const int n0 = blockIdx.x * 16;
  const int tid = threadIdx.x;
  for (int i = tid; i < 16 * 64; i += 128) xn[i] = f2bf(node[n0 * 64 + i]);
  __syncthreads();

  const int w = tid >> 5, l = tid & 31;
  const int m = l & 15, g = l >> 4;
  const int n = w * 16 + m;
  v8f acc = {};
#pragma unroll
  for (int kc = 0; kc < 2; ++kc) {
    const int xoff = kc * 32;
    BV A, B;
    const unsigned short* ar = &xn[m * 64 + xoff];
#pragma unroll
    for (int i = 0; i < 8; ++i) {
      A.u[i]     = ar[g * 8 + i];
      A.u[8 + i] = ar[16 + g * 8 + i];
    }
    const uint4* bp = (const uint4*)&rootc[n * 64 + xoff + g * 8];
    B.q[0] = bp[0]; B.q[1] = bp[2];
    acc = wmma_bf16(A, B, acc);
  }
#pragma unroll
  for (int r = 0; r < 8; ++r) {
    const int M = r + 8 * g;
    const int nid = n0 + M;
    float d = deg[nid]; d = d > 1.0f ? d : 1.0f;
    ob[M * 64 + n] = acc[r] + agg[nid * HD + n] / d + bias[n] + node[nid * HD + n];
  }
  __syncthreads();
  if (tid < 16) {
    float s = 0.0f, s2 = 0.0f;
    for (int i = 0; i < 64; ++i) { float v = ob[tid * 64 + i]; s += v; s2 += v * v; }
    float mu = s * (1.0f / 64.0f);
    float var = s2 * (1.0f / 64.0f) - mu * mu;
    mu_s[tid] = mu;
    rs_s[tid] = rsqrtf(var + LNEPS);
  }
  __syncthreads();
  for (int i = tid; i < 16 * 64; i += 128) {
    int r = i >> 6, col = i & 63;
    out[(n0 + r) * 64 + col] = (ob[i] - mu_s[r]) * rs_s[r] * gamma[col] + beta[col];
  }
}

} // anonymous namespace

extern "C" void kernel_launch(void* const* d_in, const int* in_sizes, int n_in,
                              void* d_out, int out_size, void* d_ws, size_t ws_size,
                              hipStream_t stream) {
  const float* node_emb = (const float*)d_in[0];
  const float* edge_emb = (const float*)d_in[1];
  const float* W1   = (const float*)d_in[2];
  const float* b1   = (const float*)d_in[3];
  const float* W2   = (const float*)d_in[4];
  const float* b2   = (const float*)d_in[5];
  const float* root = (const float*)d_in[6];
  const float* bias = (const float*)d_in[7];
  const float* gamma = (const float*)d_in[8];
  const float* beta  = (const float*)d_in[9];
  const long long* eidx = (const long long*)d_in[10];
  const long long* src = eidx;
  const long long* dst = eidx + NE;
  float* out = (float*)d_out;

  char* ws = (char*)d_ws;
  size_t off = 0;
  auto alloc = [&](size_t bytes) -> char* {
    char* p = ws + off;
    off = (off + bytes + 255) & ~(size_t)255;
    return p;
  };
  unsigned short* hr    = (unsigned short*)alloc((size_t)NE * 64 * 2);  // 8 MB
  unsigned short* Xsrc  = (unsigned short*)alloc((size_t)NE * 64 * 2);  // 8 MB
  unsigned short* Bbuf  = (unsigned short*)alloc((size_t)4096 * 64 * 2);// 512 KB
  unsigned short* Bbias = (unsigned short*)alloc((size_t)64 * 64 * 2);
  unsigned short* W1c   = (unsigned short*)alloc((size_t)64 * 64 * 2);
  unsigned short* rootc = (unsigned short*)alloc((size_t)64 * 64 * 2);
  float* agg = (float*)alloc((size_t)NN * 64 * 4);                      // 8 MB
  float* deg = (float*)alloc((size_t)NN * 4);                           // 128 KB

  k_zero<<<(NN * HD + 255) / 256, 256, 0, stream>>>(agg, deg);
  k_prep_w2<<<(64 * 64 * 64) / 256, 256, 0, stream>>>(W2, Bbuf);
  k_prep_small<<<4096 / 256, 256, 0, stream>>>(W1, b2, root, W1c, Bbias, rootc);
  k_gather<<<(NE * 64) / 256, 256, 0, stream>>>(node_emb, src, Xsrc);
  k_deg<<<NE / 256, 256, 0, stream>>>(dst, deg);
  k_hr<<<NE / 16, 128, 0, stream>>>(edge_emb, W1c, b1, hr);
  k_msg<<<NE / 64, 128, 0, stream>>>(hr, Xsrc, Bbuf, Bbias, dst, agg);
  k_out<<<NN / 16, 128, 0, stream>>>(node_emb, rootc, bias, gamma, beta, agg, deg, out);
}
